// textCNN_3582002725531
// MI455X (gfx1250) — matrix operand order repaired and checked by reading, not compiled, not run on hardware
//
#include <hip/hip_runtime.h>

// ---------------------------------------------------------------------------
// TextCNN on gfx1250:
//   gather+cvt(emb[x]) -> bf16 panel  (per 64-batch chunk, L2 resident)
//   TDM-staged bf16 WMMA GEMM  [rows x 320] x [320 x 448]  (f32 accum)
//   shifted-window max-pool -> tiny FC
// ---------------------------------------------------------------------------

#define BATCH    256
#define LSEQ     512
#define DEMB     300
#define KPAD     320                 // D padded to multiple of 32 (WMMA K-step)
#define NFREAL   420                 // sum_K 2*K filter rows
#define NFPAD    448                 // padded to 28 N-tiles
#define NCLS     11
#define CHUNK_B  64                  // batch rows per pass
#define CHUNK_ROWS (CHUNK_B * LSEQ)  // 32768
#define NGROUPS  7                   // 28 N-tiles / 4 per wave

#ifndef USE_TDM
#define USE_TDM 1                    // stage B panel via tensor_load_to_lds
#endif

typedef __attribute__((ext_vector_type(16))) __bf16       v16bf;
typedef __attribute__((ext_vector_type(8)))  float        v8f;
typedef __attribute__((ext_vector_type(4)))  unsigned int v4u;
typedef __attribute__((ext_vector_type(8)))  int          v8i;
typedef __attribute__((ext_vector_type(4)))  int          v4i;

union ABf16 { unsigned short u[16]; uint4 q[2]; v16bf v; };
union Pack8 { unsigned short u[8];  uint4 q; };

__device__ __forceinline__ unsigned short f2bf(float f) {
    unsigned int u = __float_as_uint(f);
    u += 0x7FFFu + ((u >> 16) & 1u);          // round-to-nearest-even
    return (unsigned short)(u >> 16);
}

// ---------------------------------------------------------------------------
// Pack conv_w[K] (2,1,K,300) into bf16 weight matrix Wb[448][320].
// Filter-row id r = K*(K-1) + c*K + j  (feats concat order f = 2*(K-1)+c).
// ---------------------------------------------------------------------------
__global__ void pack_w_kernel(const float* __restrict__ w,
                              const float* __restrict__ bias,
                              unsigned short* __restrict__ Wb,
                              float* __restrict__ biasBuf, int K) {
    int idx = blockIdx.x * blockDim.x + threadIdx.x;
    int total = 2 * K * KPAD;
    if (idx < total) {
        int c   = idx / (K * KPAD);
        int rem = idx - c * K * KPAD;
        int j   = rem / KPAD;
        int k   = rem - j * KPAD;
        int r   = K * (K - 1) + c * K + j;
        float val = (k < DEMB) ? w[(c * K + j) * DEMB + k] : 0.0f;
        Wb[(size_t)r * KPAD + k] = f2bf(val);
    }
    if (idx < 2) biasBuf[2 * (K - 1) + idx] = bias[idx];
}

__global__ void zero_pad_rows_kernel(unsigned short* __restrict__ Wb) {
    int idx = blockIdx.x * blockDim.x + threadIdx.x;
    int total = (NFPAD - NFREAL) * KPAD;     // 8960
    if (idx < total) Wb[(size_t)NFREAL * KPAD + idx] = 0;
}

// ---------------------------------------------------------------------------
// Gather + f32->bf16 convert: A[row][0..319] = bf16(emb[x[rowBase+row], :]),
// zero-padded 300->320.  8 elements per thread; aligned float4 reads
// (emb rows are 1200 B = 16 B aligned), one b128 store.
// ---------------------------------------------------------------------------
__global__ void __launch_bounds__(256)
gather_kernel(const int* __restrict__ x, const float* __restrict__ emb,
              unsigned short* __restrict__ A, int rowBase) {
    int idx = blockIdx.x * 256 + threadIdx.x;
    int row = idx / 40;                       // 40 groups of 8 per row
    int g   = idx - row * 40;
    if (row >= CHUNK_ROWS) return;
    int k0  = g * 8;
    const int    token = x[rowBase + row];
    const float* e     = emb + (size_t)token * DEMB;
    Pack8 o;
    if (k0 + 8 <= DEMB) {                     // fast path: aligned float4 pair
        float4 f0 = *(const float4*)(e + k0);
        float4 f1 = *(const float4*)(e + k0 + 4);
        o.u[0]=f2bf(f0.x); o.u[1]=f2bf(f0.y); o.u[2]=f2bf(f0.z); o.u[3]=f2bf(f0.w);
        o.u[4]=f2bf(f1.x); o.u[5]=f2bf(f1.y); o.u[6]=f2bf(f1.z); o.u[7]=f2bf(f1.w);
    } else {                                  // boundary groups (k0 >= 296)
        #pragma unroll
        for (int i = 0; i < 8; ++i) {
            int k = k0 + i;
            o.u[i] = (k < DEMB) ? f2bf(e[k]) : (unsigned short)0;
        }
    }
    *(uint4*)(A + (size_t)row * KPAD + k0) = o.q;
}

// ---------------------------------------------------------------------------
// GEMM: S[row, r] = dot(A[row,:], Wb[r,:])  in bf16 WMMA, f32 accumulate.
// Block = 8 waves, all sharing one 64-row B panel (ng selects 4 N-tiles);
// wave w computes the 16x64 strip for mtile = mtBlock*8 + w.
// B panel staged into LDS by the Tensor Data Mover (one D# issued by wave 0).
// ---------------------------------------------------------------------------
__global__ void __launch_bounds__(256)
textcnn_gemm_kernel(const unsigned short* __restrict__ A,
                    const unsigned short* __restrict__ Wb,
                    float* __restrict__ S) {
    __shared__ unsigned short Bs[64 * KPAD];          // 40 KB bf16 panel

    const int lane = threadIdx.x & 31;
    const int wave = threadIdx.x >> 5;
    const int ng   = blockIdx.x % NGROUPS;            // group of 4 N-tiles
    const int mt   = (blockIdx.x / NGROUPS) * 8 + wave;
    const int hi   = lane >> 4;                       // half-wave select
    const int m    = lane & 15;                       // A row / B col / D col

#if USE_TDM
    // ---- TDM: one tensor_load_to_lds moves the 64x320 bf16 panel (40960 B).
    if (wave == 0) {
        unsigned long long ga =
            (unsigned long long)(uintptr_t)(Wb + (size_t)ng * 64 * KPAD);
        unsigned int lds = (unsigned int)(uintptr_t)(&Bs[0]);
        const unsigned int ELEMS = 64u * KPAD;        // 20480 x 2-byte elements

        v4u g0;
        g0.x = 1u;                                    // count=1, user mode
        g0.y = lds;                                   // lds_addr
        g0.z = (unsigned int)ga;                      // global_addr[31:0]
        g0.w = (unsigned int)((ga >> 32) & 0x1FFFFFFull) | (2u << 30); // type=2

        v8i g1;
        g1[0] = (int)(1u << 16);                      // data_size=1 (2 bytes)
        g1[1] = (int)((ELEMS & 0xFFFFu) << 16);       // tensor_dim0[15:0]
        g1[2] = (int)(((ELEMS >> 16) & 0xFFFFu) | (1u << 16)); // dim0 hi, dim1=1
        g1[3] = (int)(ELEMS << 16);                   // tile_dim0 = 20480 (mult of 4B)
        g1[4] = 1;                                    // tile_dim1 = 1
        g1[5] = (int)ELEMS;                           // dim0 stride (unused, 1 row)
        g1[6] = 0;
        g1[7] = 0;

        v4i z4 = {0, 0, 0, 0};
#if defined(__clang_major__) && __clang_major__ >= 23
        v8i z8 = {0, 0, 0, 0, 0, 0, 0, 0};
        __builtin_amdgcn_tensor_load_to_lds(g0, g1, z4, z4, z8, 0);
#else
        __builtin_amdgcn_tensor_load_to_lds(g0, g1, z4, z4, 0);
#endif
        __builtin_amdgcn_s_wait_tensorcnt((short)0);
    }
    __syncthreads();
#else
    // ---- Fallback: cooperative b128 copy of the panel.
    {
        const uint4* srcB = (const uint4*)(Wb + (size_t)ng * 64 * KPAD);
        uint4*       dstB = (uint4*)Bs;
        #pragma unroll
        for (int i = 0; i < (64 * KPAD / 8) / 256; ++i)
            dstB[i * 256 + threadIdx.x] = srcB[i * 256 + threadIdx.x];
        __syncthreads();
    }
#endif

    // A fragment base: lane -> row mt*16+m, 16 contiguous K at kb+hi*16
    const unsigned short* aPtr  = A  + (size_t)(mt * 16 + m) * KPAD + hi * 16;
    // B fragment base in LDS: lane -> column m, 16 contiguous K at hi*16
    const unsigned short* bPtr0 = Bs + (size_t)( 0 + m) * KPAD + hi * 16;
    const unsigned short* bPtr1 = Bs + (size_t)(16 + m) * KPAD + hi * 16;
    const unsigned short* bPtr2 = Bs + (size_t)(32 + m) * KPAD + hi * 16;
    const unsigned short* bPtr3 = Bs + (size_t)(48 + m) * KPAD + hi * 16;

    v8f acc0 = {}, acc1 = {}, acc2 = {}, acc3 = {};

    #pragma unroll
    for (int kb = 0; kb < KPAD; kb += 32) {
        ABf16 a;
        a.q[0] = *(const uint4*)(aPtr + kb);          // K kb+hi*16 .. +7
        a.q[1] = *(const uint4*)(aPtr + kb + 8);      // K kb+hi*16+8 .. +15

        ABf16 b0, b1, b2, b3;
        b0.q[0] = *(const uint4*)(bPtr0 + kb);  b0.q[1] = *(const uint4*)(bPtr0 + kb + 8);
        b1.q[0] = *(const uint4*)(bPtr1 + kb);  b1.q[1] = *(const uint4*)(bPtr1 + kb + 8);
        b2.q[0] = *(const uint4*)(bPtr2 + kb);  b2.q[1] = *(const uint4*)(bPtr2 + kb + 8);
        b3.q[0] = *(const uint4*)(bPtr3 + kb);  b3.q[1] = *(const uint4*)(bPtr3 + kb + 8);

        acc0 = __builtin_amdgcn_wmma_f32_16x16x32_bf16(false, a.v, false, b0.v, (short)0, acc0, false, false);
        acc1 = __builtin_amdgcn_wmma_f32_16x16x32_bf16(false, a.v, false, b1.v, (short)0, acc1, false, false);
        acc2 = __builtin_amdgcn_wmma_f32_16x16x32_bf16(false, a.v, false, b2.v, (short)0, acc2, false, false);
        acc3 = __builtin_amdgcn_wmma_f32_16x16x32_bf16(false, a.v, false, b3.v, (short)0, acc3, false, false);
    }

    // D layout: lane half selects rows M+8; element e = row.
    float* sOut = S + (size_t)(mt * 16 + hi * 8) * NFPAD + ng * 64 + m;
    #pragma unroll
    for (int e = 0; e < 8; ++e) {
        sOut[(size_t)e * NFPAD +  0] = acc0[e];
        sOut[(size_t)e * NFPAD + 16] = acc1[e];
        sOut[(size_t)e * NFPAD + 32] = acc2[e];
        sOut[(size_t)e * NFPAD + 48] = acc3[e];
    }
}

// ---------------------------------------------------------------------------
// Max-pool: one wave per (batch row, feature f = 2*(K-1)+c).
// y = max_t relu(bias + sum_j S[b, t+j, rbase+j])  (shifted diagonal sum)
// ---------------------------------------------------------------------------
__global__ void __launch_bounds__(256)
pool_kernel(const float* __restrict__ S, const float* __restrict__ biasBuf,
            float* __restrict__ feats, int bBase) {
    const int lane = threadIdx.x & 31;
    const int wave = threadIdx.x >> 5;
    const int wid  = blockIdx.x * 8 + wave;
    if (wid >= CHUNK_B * 40) return;

    const int bLocal = wid / 40;
    const int f      = wid - bLocal * 40;
    const int K      = (f >> 1) + 1;
    const int c      = f & 1;
    const int rbase  = K * (K - 1) + c * K;
    const float bias = biasBuf[f];

    const float* Sb = S + (size_t)bLocal * LSEQ * NFPAD;
    float vmax = 0.0f;                               // relu lower bound
    for (int t = lane; t <= LSEQ - K; t += 32) {
        float s = bias;
        for (int j = 0; j < K; ++j)
            s += Sb[(size_t)(t + j) * NFPAD + rbase + j];
        vmax = fmaxf(vmax, s);
    }
    for (int off = 16; off > 0; off >>= 1)           // wave32 reduction
        vmax = fmaxf(vmax, __shfl_xor(vmax, off, 32));
    if (lane == 0) feats[(size_t)(bBase + bLocal) * 40 + f] = vmax;
}

// ---------------------------------------------------------------------------
// FC: out[b, n] = fc_b[n] + feats[b, :] . fc_w[n, :]
// ---------------------------------------------------------------------------
__global__ void fc_kernel(const float* __restrict__ feats,
                          const float* __restrict__ fcW,
                          const float* __restrict__ fcB,
                          float* __restrict__ out) {
    const int b = threadIdx.x;   // 256
    const int n = blockIdx.x;    // 11
    float s = fcB[n];
    #pragma unroll
    for (int f = 0; f < 40; ++f)
        s += feats[b * 40 + f] * fcW[n * 40 + f];
    out[b * NCLS + n] = s;
}

// ---------------------------------------------------------------------------
extern "C" void kernel_launch(void* const* d_in, const int* in_sizes, int n_in,
                              void* d_out, int out_size, void* d_ws, size_t ws_size,
                              hipStream_t stream) {
    (void)in_sizes; (void)n_in; (void)out_size; (void)ws_size;

    const int*   x   = (const int*)d_in[0];
    const float* emb = (const float*)d_in[1];
    // d_in[2..21] = conv_w[0..19], d_in[22..41] = conv_b[0..19]
    const float* fcW = (const float*)d_in[42];
    const float* fcB = (const float*)d_in[43];
    float* out = (float*)d_out;

    // Workspace layout (~80.1 MB total)
    char* ws = (char*)d_ws;
    unsigned short* Wb      = (unsigned short*)ws;               // 286,720 B
    float*          biasBuf = (float*)(ws + 286720);             // 160 B
    float*          feats   = (float*)(ws + 286880);             // 40,960 B
    unsigned short* Abuf    = (unsigned short*)(ws + 327936);    // 20,971,520 B
    float*          S       = (float*)(ws + 21299456);           // 58,720,256 B

    // 1) Weights -> bf16 B-matrix + bias gather.
    zero_pad_rows_kernel<<<(8960 + 255) / 256, 256, 0, stream>>>(Wb);
    for (int i = 0; i < 20; ++i) {
        const int K = i + 1;
        const int total = 2 * K * KPAD;
        pack_w_kernel<<<(total + 255) / 256, 256, 0, stream>>>(
            (const float*)d_in[2 + i], (const float*)d_in[22 + i], Wb, biasBuf, K);
    }

    // 2) Chunked gather -> GEMM -> pooling (stream order gives dependencies).
    const int gatherBlocks = (CHUNK_ROWS * 40 + 255) / 256;      // 5120
    const int gemmBlocks   = (CHUNK_ROWS / 16 / 8) * NGROUPS;    // 1792
    const int poolBlocks   = (CHUNK_B * 40 + 7) / 8;             // 320
    for (int cb = 0; cb < BATCH; cb += CHUNK_B) {
        gather_kernel<<<gatherBlocks, 256, 0, stream>>>(x, emb, Abuf, cb * LSEQ);
        textcnn_gemm_kernel<<<gemmBlocks, 256, 0, stream>>>(Abuf, Wb, S);
        pool_kernel<<<poolBlocks, 256, 0, stream>>>(S, biasBuf, feats, cb);
    }

    // 3) FC head.
    fc_kernel<<<NCLS, 256, 0, stream>>>(feats, fcW, fcB, out);
}
